// CNNLSTM_14869176778812
// MI455X (gfx1250) — matrix operand order repaired and checked by reading, not compiled, not run on hardware
//
#include <hip/hip_runtime.h>
#include <hip/hip_bf16.h>

// CNN(3x conv1d k=3) + MaxPool + 2-layer LSTM + masked mean + FC head.
// All GEMM-shaped work on v_wmma_f32_16x16x32_f16 (f16 in, f32 acc).
// Activations kept in [B, T, C] (channel-minor) f16 layout so every WMMA
// fragment load is a contiguous 16/32-byte vector load.
// Big GEMMs / convs use 2x2 register tiling: 4 WMMA per 8 b128 fragment loads.

typedef __attribute__((ext_vector_type(16))) _Float16 v16h;
typedef __attribute__((ext_vector_type(8)))  _Float16 v8h;
typedef __attribute__((ext_vector_type(8)))  float    v8f;

__device__ __forceinline__ v8f wmma16(v16h a, v16h b, v8f c) {
    return __builtin_amdgcn_wmma_f32_16x16x32_f16(
        false, a, false, b, (short)0, c, false, false);
}

__device__ __forceinline__ float fast_sigmoid(float x) {
    x = fminf(fmaxf(x, -30.f), 30.f);
    return 1.f / (1.f + __expf(-x));
}
__device__ __forceinline__ float fast_tanh(float x) {
    x = fminf(fmaxf(x, -15.f), 15.f);
    float e = __expf(2.f * x);
    return (e - 1.f) / (e + 1.f);
}

// A-fragment: two contiguous 16B loads per lane (row = lane&15, K split by hi)
__device__ __forceinline__ v16h load_a16(const _Float16* p) {
    union { v16h v; v8h h[2]; } u;
    u.h[0] = *(const v8h*)(p);
    u.h[1] = *(const v8h*)(p + 8);
    return u.v;
}

// ---------------------------------------------------------------- converts
// x [64,1024,512] f32 (NCT) -> xt [64,512,1024] f16 (NTC)
__global__ void cvt_x_btc(const float* __restrict__ x, _Float16* __restrict__ xt) {
    int idx = blockIdx.x * blockDim.x + threadIdx.x;
    const int total = 64 * 512 * 1024;
    if (idx >= total) return;
    int c = idx & 1023;
    int t = (idx >> 10) & 511;
    int b = idx >> 19;
    xt[idx] = (_Float16)x[((size_t)b * 1024 + c) * 512 + t];
}

__global__ void f32_to_f16_kernel(const float* __restrict__ in,
                                  _Float16* __restrict__ out, int n) {
    int i = blockIdx.x * blockDim.x + threadIdx.x;
    if (i < n) out[i] = (_Float16)in[i];
}

// conv weights [O,I,3] f32 -> [3][O][I] f16 (K-contiguous A operand)
__global__ void repack_conv_w(const float* __restrict__ w,
                              _Float16* __restrict__ wp, int O, int I) {
    int idx = blockIdx.x * blockDim.x + threadIdx.x;
    if (idx >= O * I * 3) return;
    int i  = idx % I;
    int o  = (idx / I) % O;
    int kk = idx / (I * O);
    wp[idx] = (_Float16)w[((size_t)o * I + i) * 3 + kk];
}

// ------------------------------------------ fast GEMM: guard-free, 2x2 tiles
// C[M,N] = act(A[M,K] x Bt[N,K]^T + bias); requires M%32==0, N%32==0, K%32==0.
__global__ void gemm_nt_wmma_fast(const _Float16* __restrict__ A,
                                  const _Float16* __restrict__ Bt,
                                  const float* __restrict__ bias,
                                  _Float16* __restrict__ out16,
                                  int M, int N, int K, int relu) {
    const int nt2 = N >> 5;
    const int wave = blockIdx.x * (blockDim.x >> 5) + (threadIdx.x >> 5);
    if (wave >= (M >> 5) * nt2) return;   // wave-uniform
    const int m0 = (wave / nt2) << 5;
    const int n0 = (wave % nt2) << 5;
    const int l = threadIdx.x & 31, lm = l & 15, hi = l >> 4;
    const _Float16* ap0 = A  + (size_t)(m0 + lm) * K + hi * 16;
    const _Float16* ap1 = ap0 + (size_t)16 * K;
    const _Float16* bp0 = Bt + (size_t)(n0 + lm) * K + hi * 16;
    const _Float16* bp1 = bp0 + (size_t)16 * K;
    v8f acc00 = {}, acc01 = {}, acc10 = {}, acc11 = {};
#pragma unroll 2
    for (int k0 = 0; k0 < K; k0 += 32) {
        __builtin_prefetch(bp0 + k0 + 256, 0, 1);
        v16h a0 = load_a16(ap0 + k0);
        v16h a1 = load_a16(ap1 + k0);
        v16h b0 = *(const v16h*)(bp0 + k0);
        v16h b1 = *(const v16h*)(bp1 + k0);
        acc00 = wmma16(a0, b0, acc00);
        acc01 = wmma16(a0, b1, acc01);
        acc10 = wmma16(a1, b0, acc10);
        acc11 = wmma16(a1, b1, acc11);
    }
    const float bv0 = bias ? bias[n0 + lm] : 0.f;
    const float bv1 = bias ? bias[n0 + 16 + lm] : 0.f;
#pragma unroll
    for (int r = 0; r < 8; ++r) {
        int row0 = m0 + r + 8 * hi;
        int row1 = row0 + 16;
        float v00 = acc00[r] + bv0, v01 = acc01[r] + bv1;
        float v10 = acc10[r] + bv0, v11 = acc11[r] + bv1;
        if (relu) {
            v00 = fmaxf(v00, 0.f); v01 = fmaxf(v01, 0.f);
            v10 = fmaxf(v10, 0.f); v11 = fmaxf(v11, 0.f);
        }
        out16[(size_t)row0 * N + n0 + lm]      = (_Float16)v00;
        out16[(size_t)row0 * N + n0 + 16 + lm] = (_Float16)v01;
        out16[(size_t)row1 * N + n0 + lm]      = (_Float16)v10;
        out16[(size_t)row1 * N + n0 + 16 + lm] = (_Float16)v11;
    }
}

// ------------------------------------- generic guarded GEMM (tiny FC tails)
__global__ void gemm_nt_wmma(const _Float16* __restrict__ A,
                             const _Float16* __restrict__ Bt,
                             const float* __restrict__ bias,
                             _Float16* __restrict__ out16,
                             float* __restrict__ out32,
                             int M, int N, int K, int relu) {
    const int mt = (M + 15) >> 4, nt = (N + 15) >> 4;
    const int wave = blockIdx.x * (blockDim.x >> 5) + (threadIdx.x >> 5);
    if (wave >= mt * nt) return;
    const int m0 = (wave / nt) * 16;
    const int n0 = (wave % nt) * 16;
    const int l = threadIdx.x & 31, lm = l & 15, hi = l >> 4;
    v8f acc = {};
    for (int k0 = 0; k0 < K; k0 += 32) {
        v16h a, b;
#pragma unroll
        for (int e = 0; e < 16; ++e) {
            int ka = k0 + hi * 16 + e;
            a[e] = (m0 + lm < M && ka < K) ? A[(size_t)(m0 + lm) * K + ka]
                                           : (_Float16)0.f;
            int kb = k0 + hi * 16 + e;
            b[e] = (n0 + lm < N && kb < K) ? Bt[(size_t)(n0 + lm) * K + kb]
                                           : (_Float16)0.f;
        }
        acc = wmma16(a, b, acc);
    }
    const int col = n0 + lm;
    const float bv = (bias && col < N) ? bias[col] : 0.f;
#pragma unroll
    for (int r = 0; r < 8; ++r) {
        int row = m0 + r + 8 * hi;
        if (row < M && col < N) {
            float v = acc[r] + bv;
            if (relu) v = fmaxf(v, 0.f);
            if (out16) out16[(size_t)row * N + col] = (_Float16)v;
            if (out32) out32[(size_t)row * N + col] = v;
        }
    }
}

// --------------------------------------- conv1d k=3 pad=1, BTC implicit GEMM
// X:[64,T,I] f16   Wp:[3][O][I] f16   Y:[64,T,O] f16  (bias+ReLU fused)
// One wave computes a 32(O) x 32(t) tile: 4 WMMA per 8 b128 fragment loads.
__global__ void conv1d_k3_btc_wmma(const _Float16* __restrict__ X,
                                   const _Float16* __restrict__ Wp,
                                   const float* __restrict__ bias,
                                   _Float16* __restrict__ Y,
                                   int I, int O) {
    const int T = 512;
    const int nt = (64 * T) / 32;            // 1024 double-column tiles
    const int wave = blockIdx.x * (blockDim.x >> 5) + (threadIdx.x >> 5);
    if (wave >= (O / 32) * nt) return;
    const int m0 = (wave / nt) * 32;
    const int c0 = (wave % nt) * 32;
    const int b  = c0 >> 9;                  // T == 512
    const int t0 = c0 & 511;
    const int l = threadIdx.x & 31, lm = l & 15, hi = l >> 4;
    v8f acc00 = {}, acc01 = {}, acc10 = {}, acc11 = {};
    const v16h zero16 = {};
#pragma unroll
    for (int kk = 0; kk < 3; ++kk) {
        const int tp0 = t0 + lm + kk - 1;    // per-lane shifted column (tile 0)
        const int tp1 = tp0 + 16;            // tile 1
        const bool v0 = (tp0 >= 0) && (tp0 < T);
        const bool v1 = (tp1 >= 0) && (tp1 < T);
        const _Float16* ap0 = Wp + (size_t)(kk * O + m0 + lm) * I + hi * 16;
        const _Float16* ap1 = ap0 + (size_t)16 * I;
        const _Float16* x0 = X + ((size_t)b * T + tp0) * I + hi * 16;
        const _Float16* x1 = X + ((size_t)b * T + tp1) * I + hi * 16;
#pragma unroll 2
        for (int k0 = 0; k0 < I; k0 += 32) {
            v16h a0 = load_a16(ap0 + k0);
            v16h a1 = load_a16(ap1 + k0);
            v16h b0 = v0 ? *(const v16h*)(x0 + k0) : zero16;
            v16h b1 = v1 ? *(const v16h*)(x1 + k0) : zero16;
            acc00 = wmma16(a0, b0, acc00);
            acc01 = wmma16(a0, b1, acc01);
            acc10 = wmma16(a1, b0, acc10);
            acc11 = wmma16(a1, b1, acc11);
        }
    }
    // epilogue: contiguous output channels per lane -> 16B vector stores
    float bv0[8], bv1[8];
#pragma unroll
    for (int r = 0; r < 8; ++r) {
        bv0[r] = bias[m0 + 8 * hi + r];
        bv1[r] = bias[m0 + 16 + 8 * hi + r];
    }
    v8h o00, o01, o10, o11;
#pragma unroll
    for (int r = 0; r < 8; ++r) {
        o00[r] = (_Float16)fmaxf(acc00[r] + bv0[r], 0.f);
        o01[r] = (_Float16)fmaxf(acc01[r] + bv0[r], 0.f);
        o10[r] = (_Float16)fmaxf(acc10[r] + bv1[r], 0.f);
        o11[r] = (_Float16)fmaxf(acc11[r] + bv1[r], 0.f);
    }
    _Float16* y0 = Y + ((size_t)b * T + t0 + lm) * O + m0 + 8 * hi;  // tile (0,0)
    *(v8h*)(y0)                       = o00;
    *(v8h*)(y0 + (size_t)16 * O)      = o01;          // t tile 1
    *(v8h*)(y0 + 16)                  = o10;          // O tile 1
    *(v8h*)(y0 + (size_t)16 * O + 16) = o11;
}

// --------------------------------- maxpool(2) over t, BTC in -> BTC out
__global__ void maxpool_btc(const _Float16* __restrict__ In,  // [64,512,192]
                            _Float16* __restrict__ Out) {     // [64,256,192]
    int idx = blockIdx.x * blockDim.x + threadIdx.x;
    const int total = 64 * 256 * 192;
    if (idx >= total) return;
    int c  = idx % 192;
    int t2 = (idx / 192) % 256;
    int b  = idx / (192 * 256);
    const _Float16* p = In + ((size_t)b * 512 + 2 * t2) * 192 + c;
    Out[idx] = (_Float16)fmaxf((float)p[0], (float)p[192]);
}

// ------------------------------------------------------- persistent LSTM layer
// 4 workgroups (16 batch rows) x 32 waves (16 h-cols). Four gate fragments +
// cell state + masked-mean pooling live in VGPRs across all 256 steps;
// h double-buffered in LDS, one barrier per timestep.
__global__ void __launch_bounds__(1024) lstm_layer_wmma(
        const _Float16* __restrict__ XP,   // [64,256,2048] = x@W_ih^T + b
        const _Float16* __restrict__ Whh,  // [2048,512] f16
        _Float16* __restrict__ HSeq,       // [64,256,512] f16 or null
        float* __restrict__ Pooled,        // [64,512] f32 or null
        const int* __restrict__ Avalid) {  // [64] or null
    constexpr int H = 512, T2 = 256, G = 2048;
    __shared__ __attribute__((aligned(32))) _Float16 hbuf[2][16 * H];
    const int b0 = blockIdx.x * 16;
    const int w  = threadIdx.x >> 5;
    const int j0 = w * 16;
    const int l  = threadIdx.x & 31, lm = l & 15, hi = l >> 4;

    for (int i = threadIdx.x; i < 16 * H; i += blockDim.x) hbuf[0][i] = (_Float16)0.f;
    __syncthreads();

    int av[8];
#pragma unroll
    for (int r = 0; r < 8; ++r)
        av[r] = Avalid ? Avalid[b0 + r + 8 * hi] : 1;

    v8f cfrag = {};
    v8f pool  = {};

    for (int t = 0; t < T2; ++t) {
        const int cur = t & 1, nxt = cur ^ 1;
        v8f g[4];
#pragma unroll
        for (int gi = 0; gi < 4; ++gi) {
#pragma unroll
            for (int r = 0; r < 8; ++r) {
                int bb = b0 + r + 8 * hi;
                g[gi][r] = (float)XP[((size_t)bb * T2 + t) * G + gi * H + j0 + lm];
            }
        }
        for (int k0 = 0; k0 < H; k0 += 32) {
            const _Float16* hp = &hbuf[cur][lm * H + k0 + hi * 16];
            union { v16h v; v8h h8[2]; } ua;
            ua.h8[0] = *(const v8h*)(hp);
            ua.h8[1] = *(const v8h*)(hp + 8);
            __builtin_prefetch(Whh + (size_t)(j0 + lm) * H + k0 + 32, 0, 1);
#pragma unroll
            for (int gi = 0; gi < 4; ++gi) {
                v16h bf = *(const v16h*)(Whh + (size_t)(gi * H + j0 + lm) * H +
                                         k0 + hi * 16);
                g[gi] = wmma16(ua.v, bf, g[gi]);
            }
        }
#pragma unroll
        for (int r = 0; r < 8; ++r) {
            float ig = fast_sigmoid(g[0][r]);
            float fg = fast_sigmoid(g[1][r]);
            float gg = fast_tanh(g[2][r]);
            float og = fast_sigmoid(g[3][r]);
            float c  = fg * cfrag[r] + ig * gg;
            float h  = og * fast_tanh(c);
            cfrag[r] = c;
            if (Pooled && t < av[r]) pool[r] += h;
            int bb = b0 + r + 8 * hi;
            if (HSeq) HSeq[((size_t)bb * T2 + t) * H + j0 + lm] = (_Float16)h;
            hbuf[nxt][(r + 8 * hi) * H + j0 + lm] = (_Float16)h;
        }
        __syncthreads();
    }
    if (Pooled) {
#pragma unroll
        for (int r = 0; r < 8; ++r)
            Pooled[(size_t)(b0 + r + 8 * hi) * H + j0 + lm] = pool[r] / (float)av[r];
    }
}

// ===========================================================================
extern "C" void kernel_launch(void* const* d_in, const int* in_sizes, int n_in,
                              void* d_out, int out_size, void* d_ws, size_t ws_size,
                              hipStream_t stream) {
    (void)in_sizes; (void)n_in; (void)out_size; (void)ws_size;
    const float* x    = (const float*)d_in[0];
    const int*   a    = (const int*)d_in[1];
    const float* c1w  = (const float*)d_in[2];  const float* c1b = (const float*)d_in[3];
    const float* c2w  = (const float*)d_in[4];  const float* c2b = (const float*)d_in[5];
    const float* c3w  = (const float*)d_in[6];  const float* c3b = (const float*)d_in[7];
    const float* wih0 = (const float*)d_in[8];  const float* whh0 = (const float*)d_in[9];
    const float* b0   = (const float*)d_in[10];
    const float* wih1 = (const float*)d_in[11]; const float* whh1 = (const float*)d_in[12];
    const float* b1   = (const float*)d_in[13];
    const float* f1w  = (const float*)d_in[14]; const float* f1b = (const float*)d_in[15];
    const float* f2w  = (const float*)d_in[16]; const float* f2b = (const float*)d_in[17];
    const float* f3w  = (const float*)d_in[18]; const float* f3b = (const float*)d_in[19];
    const float* f4w  = (const float*)d_in[20]; const float* f4b = (const float*)d_in[21];
    float* out = (float*)d_out;

    constexpr int B = 64, T = 512, T2 = 256, H = 512, G = 2048;
    constexpr int C0 = 1024, C1 = 768, C2 = 384, C3 = 192;

    char* ws = (char*)d_ws;
    size_t off = 0;
    auto alloc = [&](size_t bytes) -> void* {
        void* p = ws + off;
        off = (off + bytes + 255) & ~(size_t)255;
        return p;
    };
    auto allocH = [&](size_t n) { return (_Float16*)alloc(n * 2); };

    // f16 weights
    _Float16* wc1  = allocH((size_t)C1 * C0 * 3);   // packed [3][O][I]
    _Float16* wc2  = allocH((size_t)C2 * C1 * 3);
    _Float16* wc3  = allocH((size_t)C3 * C2 * 3);
    _Float16* wI0  = allocH((size_t)G * C3);
    _Float16* wH0  = allocH((size_t)G * H);
    _Float16* wI1  = allocH((size_t)G * H);
    _Float16* wH1  = allocH((size_t)G * H);
    _Float16* wF1  = allocH((size_t)256 * H);
    _Float16* wF2  = allocH((size_t)64 * 256);
    _Float16* wF3  = allocH((size_t)8 * 64);
    _Float16* wF4  = allocH((size_t)1 * 8);
    // activation regions (aliased: regA = x16 then xp; regB = h1 then hs0)
    _Float16* regA = allocH((size_t)B * C0 * T);   // == B*T2*G elements
    _Float16* regB = allocH((size_t)B * C1 * T);
    _Float16* regC = allocH((size_t)B * C2 * T);
    _Float16* regD = allocH((size_t)B * C3 * T);
    _Float16* regE = allocH((size_t)B * T2 * C3);
    float*    pooled   = (float*)alloc((size_t)B * H * 4);
    _Float16* pooled16 = allocH((size_t)B * H);
    _Float16* z1 = allocH((size_t)B * 256);
    _Float16* z2 = allocH((size_t)B * 64);
    _Float16* z3 = allocH((size_t)B * 8);

    auto cvt = [&](const float* src, _Float16* dst, size_t n) {
        f32_to_f16_kernel<<<(int)((n + 255) / 256), 256, 0, stream>>>(src, dst, (int)n);
    };
    auto repack = [&](const float* src, _Float16* dst, int O, int I) {
        int n = O * I * 3;
        repack_conv_w<<<(n + 255) / 256, 256, 0, stream>>>(src, dst, O, I);
    };
    auto gemmF = [&](const _Float16* A, const _Float16* Bt, const float* bias,
                     _Float16* o16, int M, int N, int K, int relu) {
        int tiles = (M >> 5) * (N >> 5);
        gemm_nt_wmma_fast<<<(tiles + 3) / 4, 128, 0, stream>>>(A, Bt, bias, o16,
                                                               M, N, K, relu);
    };
    auto gemmG = [&](const _Float16* A, const _Float16* Bt, const float* bias,
                     _Float16* o16, float* o32, int M, int N, int K, int relu) {
        int tiles = ((M + 15) / 16) * ((N + 15) / 16);
        gemm_nt_wmma<<<(tiles + 3) / 4, 128, 0, stream>>>(A, Bt, bias, o16, o32,
                                                          M, N, K, relu);
    };
    auto conv = [&](const _Float16* X, const _Float16* Wp, const float* bias,
                    _Float16* Y, int I, int O) {
        int tiles = (O / 32) * (B * T / 32);
        conv1d_k3_btc_wmma<<<(tiles + 3) / 4, 128, 0, stream>>>(X, Wp, bias, Y, I, O);
    };

    // 1) input -> f16 NTC; weights -> f16 (conv packed [3][O][I])
    cvt_x_btc<<<(B * T * C0 + 255) / 256, 256, 0, stream>>>(x, regA);
    repack(c1w, wc1, C1, C0);
    repack(c2w, wc2, C2, C1);
    repack(c3w, wc3, C3, C2);
    cvt(wih0, wI0, (size_t)G * C3);
    cvt(whh0, wH0, (size_t)G * H);
    cvt(wih1, wI1, (size_t)G * H);
    cvt(whh1, wH1, (size_t)G * H);
    cvt(f1w, wF1, (size_t)256 * H);
    cvt(f2w, wF2, (size_t)64 * 256);
    cvt(f3w, wF3, (size_t)8 * 64);
    cvt(f4w, wF4, (size_t)1 * 8);

    // 2) conv stack (implicit GEMM in BTC layout, bias+ReLU fused)
    conv(regA, wc1, c1b, regB, C0, C1);
    conv(regB, wc2, c2b, regC, C1, C2);
    conv(regC, wc3, c3b, regD, C2, C3);

    // 3) maxpool(2) -> [64,256,192] (already the LSTM input layout)
    maxpool_btc<<<(B * T2 * C3 + 255) / 256, 256, 0, stream>>>(regD, regE);

    // 4) LSTM layer 0
    gemmF(regE, wI0, b0, regA /*xp*/, B * T2, G, C3, 0);
    lstm_layer_wmma<<<4, 1024, 0, stream>>>(regA, wH0, regB /*h-seq*/, nullptr, nullptr);

    // 5) LSTM layer 1 (fused masked-mean pooling)
    gemmF(regB, wI1, b1, regA /*xp*/, B * T2, G, H, 0);
    lstm_layer_wmma<<<4, 1024, 0, stream>>>(regA, wH1, nullptr, pooled, a);

    // 6) FC head (ReLU after every Linear)
    cvt(pooled, pooled16, (size_t)B * H);
    gemmF(pooled16, wF1, f1b, z1, B, 256, H, 1);
    gemmF(z1, wF2, f2b, z2, B, 64, 256, 1);
    gemmG(z2, wF3, f3b, z3, nullptr, B, 8, 64, 1);
    gemmG(z3, wF4, f4b, nullptr, out, B, 1, 8, 1);
}
